// OfficialChimeraLayer_9234179686551
// MI455X (gfx1250) — compile-verified, matrix operand order checked by reading, not run on hardware
//
#include <hip/hip_runtime.h>
#include <hip/hip_bf16.h>

typedef __bf16 bf16;
typedef __attribute__((ext_vector_type(16))) __bf16 bf16x16;
typedef __attribute__((ext_vector_type(8)))  __bf16 bf16x8;
typedef __attribute__((ext_vector_type(8)))  float  f32x8;

#define D_MODEL   1024
#define D_INNER   2048
#define CONV_DIM  2176
#define D_IN_PROJ 4256
#define NHEADS    32
#define HEADDIM   64
#define D_STATE   64
#define SEQLEN    2048
#define BATCH     4
#define ROWS      (BATCH*SEQLEN)     // 8192
#define RMS_EPS   1.1920929e-07f
#define GNORM_EPS 1e-5f

// ---------------------------------------------------------------- helpers
__device__ __forceinline__ float silu_f(float x) { return x / (1.f + __expf(-x)); }
__device__ __forceinline__ float softplus_f(float x) {
    return x > 20.f ? x : log1pf(__expf(x));
}

// Load one 16x32 bf16 fragment (A-style, or B sourced as rows of W).
// p must already point at rowStart + (laneHi ? 8 : 0); loads k..k+7 and k+16..k+23.
__device__ __forceinline__ bf16x16 ldfrag(const bf16* __restrict__ p) {
    bf16x8 lo = *reinterpret_cast<const bf16x8*>(p);
    bf16x8 hi = *reinterpret_cast<const bf16x8*>(p + 16);
    bf16x16 f;
#pragma unroll
    for (int i = 0; i < 8; i++) { f[i] = lo[i]; f[i + 8] = hi[i]; }
    return f;
}

#define WMMA_BF16(a, b, c) \
    __builtin_amdgcn_wmma_f32_16x16x32_bf16(false, (a), false, (b), (short)0, (c), false, false)

// One wave computes a 32x64 f32 tile of A[MxK] * W[NxK]^T (both bf16, row major).
// acc layout: acc[i*4+j] = rows m0+i*16.., cols n0+j*16..
__device__ __forceinline__ void mma_tile32x64(const bf16* __restrict__ A,
                                              const bf16* __restrict__ W,
                                              int K, int m0, int n0, f32x8 acc[8]) {
    const int lane = threadIdx.x & 31;
    const int r    = lane & 15;
    const int hi8  = (lane >> 4) * 8;
    const bf16* a0 = A + (size_t)(m0 + r) * K + hi8;
    const bf16* a1 = a0 + (size_t)16 * K;
    const bf16* b0 = W + (size_t)(n0 + r) * K + hi8;
    const bf16* b1 = b0 + (size_t)16 * K;
    const bf16* b2 = b0 + (size_t)32 * K;
    const bf16* b3 = b0 + (size_t)48 * K;
    for (int k = 0; k < K; k += 32) {
        __builtin_prefetch(a0 + k + 512, 0, 1);   // stream-ahead on A
        bf16x16 fa0 = ldfrag(a0 + k);
        bf16x16 fa1 = ldfrag(a1 + k);
        bf16x16 fb0 = ldfrag(b0 + k);
        bf16x16 fb1 = ldfrag(b1 + k);
        bf16x16 fb2 = ldfrag(b2 + k);
        bf16x16 fb3 = ldfrag(b3 + k);
        acc[0] = WMMA_BF16(fa0, fb0, acc[0]);
        acc[1] = WMMA_BF16(fa0, fb1, acc[1]);
        acc[2] = WMMA_BF16(fa0, fb2, acc[2]);
        acc[3] = WMMA_BF16(fa0, fb3, acc[3]);
        acc[4] = WMMA_BF16(fa1, fb0, acc[4]);
        acc[5] = WMMA_BF16(fa1, fb1, acc[5]);
        acc[6] = WMMA_BF16(fa1, fb2, acc[6]);
        acc[7] = WMMA_BF16(fa1, fb3, acc[7]);
    }
}

// ---------------------------------------------------------------- kernels
__global__ void k_f32_to_bf16(const float* __restrict__ in, bf16* __restrict__ out, int n) {
    int i = blockIdx.x * 256 + threadIdx.x;
    if (i < n) out[i] = (bf16)in[i];
}

// RMSNorm of x (one row of 1024 per block); writes fp32 + bf16 copies.
__global__ __launch_bounds__(256) void k_rmsnorm(const float* __restrict__ x,
                                                 const float* __restrict__ w,
                                                 float* __restrict__ xn,
                                                 bf16*  __restrict__ xnb) {
    const size_t row = blockIdx.x;
    const float* xr = x + row * D_MODEL;
    __shared__ float red[256];
    float ss = 0.f;
    for (int i = threadIdx.x; i < D_MODEL; i += 256) { float v = xr[i]; ss += v * v; }
    red[threadIdx.x] = ss; __syncthreads();
    for (int o = 128; o > 0; o >>= 1) {
        if (threadIdx.x < o) red[threadIdx.x] += red[threadIdx.x + o];
        __syncthreads();
    }
    float scale = rsqrtf(red[0] * (1.f / D_MODEL) + RMS_EPS);
    for (int i = threadIdx.x; i < D_MODEL; i += 256) {
        float v = xr[i] * scale * w[i];
        xn[row * D_MODEL + i]  = v;
        xnb[row * D_MODEL + i] = (bf16)v;
    }
}

// mean over S for each (b, d)
__global__ void k_meanpool(const float* __restrict__ xn, float* __restrict__ mean) {
    int b = blockIdx.y;
    int d = blockIdx.x * 256 + threadIdx.x;
    const float* p = xn + ((size_t)b * SEQLEN) * D_MODEL + d;
    float s = 0.f;
    for (int t = 0; t < SEQLEN; t++) s += p[(size_t)t * D_MODEL];
    mean[b * D_MODEL + d] = s * (1.f / SEQLEN);
}

// tiny router MLP + softmax -> (p_fast, p_mamba) per batch
__global__ void k_router(const float* __restrict__ mean,
                         const float* __restrict__ w1, const float* __restrict__ b1,
                         const float* __restrict__ w2, const float* __restrict__ b2,
                         float* __restrict__ pfpm) {
    int b = threadIdx.x >> 5;
    int j = threadIdx.x & 31;
    __shared__ float h[BATCH][32];
    const float* m  = mean + b * D_MODEL;
    const float* wr = w1 + j * D_MODEL;
    float acc = b1[j];
    for (int k = 0; k < D_MODEL; k++) acc += m[k] * wr[k];
    h[b][j] = silu_f(acc);
    __syncthreads();
    if (j == 0) {
        float l[3];
        for (int o = 0; o < 3; o++) {
            float a = b2[o];
            for (int k = 0; k < 32; k++) a += h[b][k] * w2[o * 32 + k];
            l[o] = a;
        }
        float mx = fmaxf(l[0], fmaxf(l[1], l[2]));
        float e0 = __expf(l[0] - mx), e1 = __expf(l[1] - mx), e2 = __expf(l[2] - mx);
        float inv = 1.f / (e0 + e1 + e2);
        pfpm[b * 2 + 0] = e0 * inv;
        pfpm[b * 2 + 1] = (e1 + e2) * inv;
    }
}

// zxbcdt[8192,4256] = xnorm_bf16[8192,1024] @ in_proj_w_bf16[4256,1024]^T
__global__ __launch_bounds__(256) void k_gemm_inproj(const bf16* __restrict__ A,
                                                     const bf16* __restrict__ W,
                                                     float* __restrict__ C) {
    const int wave = threadIdx.x >> 5;
    const int m0 = blockIdx.y * 256 + wave * 32;
    int n0 = blockIdx.x * 64;
    if (n0 > D_IN_PROJ - 64) n0 = D_IN_PROJ - 64;   // overlap tail tile (benign rewrite)
    f32x8 z = {0.f, 0.f, 0.f, 0.f, 0.f, 0.f, 0.f, 0.f};
    f32x8 acc[8] = {z, z, z, z, z, z, z, z};
    mma_tile32x64(A, W, D_MODEL, m0, n0, acc);
    const int lane = threadIdx.x & 31;
    const int r = lane & 15, rb = (lane >> 4) * 8;
#pragma unroll
    for (int i = 0; i < 2; i++)
#pragma unroll
        for (int j = 0; j < 4; j++) {
            float* cp = C + (size_t)(m0 + i * 16 + rb) * D_IN_PROJ + (n0 + j * 16 + r);
            f32x8 v = acc[i * 4 + j];
#pragma unroll
            for (int q = 0; q < 8; q++) cp[(size_t)q * D_IN_PROJ] = v[q];
        }
}

// causal depthwise conv(4) + bias + SiLU over the xBC slice of zxbcdt
__global__ __launch_bounds__(256) void k_conv(const float* __restrict__ zxbcdt,
                                              const float* __restrict__ cw,
                                              const float* __restrict__ cb,
                                              float* __restrict__ xbc) {
    size_t idx = (size_t)blockIdx.x * 256 + threadIdx.x;   // over 8192*2176
    int c = (int)(idx % CONV_DIM);
    size_t row = idx / CONV_DIM;                           // b*2048 + s
    int s = (int)(row & (SEQLEN - 1));
    const float* base = zxbcdt + row * D_IN_PROJ + D_INNER + c;
    float acc = cb[c];
#pragma unroll
    for (int t = 0; t < 4; t++) {
        int ss = s - 3 + t;
        if (ss >= 0) acc += cw[c * 4 + t] * base[(long)(t - 3) * D_IN_PROJ];
    }
    xbc[row * CONV_DIM + c] = silu_f(acc);
}

// sequential SSM scan: one workgroup per (b,h); 64x64 state in registers
__global__ __launch_bounds__(256) void k_scan(const float* __restrict__ zxbcdt,
                                              const float* __restrict__ xbc,
                                              const float* __restrict__ dt_bias,
                                              const float* __restrict__ A_log,
                                              const float* __restrict__ Dp,
                                              float* __restrict__ y) {
    const int b = blockIdx.x >> 5;
    const int h = blockIdx.x & 31;
    const int t = threadIdx.x;
    const int p  = t & 63;
    const int nb = t >> 6;                 // n block: nb*16 .. nb*16+15
    float st[16];
#pragma unroll
    for (int i = 0; i < 16; i++) st[i] = 0.f;
    const float A  = -__expf(A_log[h]);
    const float db = dt_bias[h];
    const float Dh = Dp[h];
    __shared__ float sB[D_STATE], sC[D_STATE], part[4][HEADDIM];
    const size_t dtcol = ((size_t)b * SEQLEN) * D_IN_PROJ + (D_INNER + CONV_DIM) + h;
    const size_t rowx0 = ((size_t)b * SEQLEN) * CONV_DIM;
    for (int s = 0; s < SEQLEN; s++) {
        const size_t rx = rowx0 + (size_t)s * CONV_DIM;
        if (t < 64)        sB[t]      = xbc[rx + D_INNER + t];
        else if (t < 128)  sC[t - 64] = xbc[rx + D_INNER + D_STATE + (t - 64)];
        float dtv = softplus_f(zxbcdt[dtcol + (size_t)s * D_IN_PROJ] + db);
        float dA  = __expf(dtv * A);
        float xp  = xbc[rx + h * HEADDIM + p];
        __syncthreads();
        float coef = dtv * xp;
        float acc = 0.f;
#pragma unroll
        for (int i = 0; i < 16; i++) {
            int n = nb * 16 + i;
            float hv = st[i] * dA + coef * sB[n];
            st[i] = hv;
            acc += hv * sC[n];
        }
        part[nb][p] = acc;
        __syncthreads();
        if (t < 64) {
            float yv = part[0][t] + part[1][t] + part[2][t] + part[3][t] + Dh * xp;
            y[((size_t)b * SEQLEN + s) * D_INNER + h * HEADDIM + t] = yv;
        }
    }
}

// gated RMSNorm: yg = rmsnorm(y * silu(z)) * gw, output bf16 for out_proj
__global__ __launch_bounds__(256) void k_gnorm(const float* __restrict__ y,
                                               const float* __restrict__ zxbcdt,
                                               const float* __restrict__ gw,
                                               bf16* __restrict__ yg) {
    const size_t row = blockIdx.x;
    const float* yr = y + row * D_INNER;
    const float* zr = zxbcdt + row * D_IN_PROJ;   // z = first 2048 cols
    __shared__ float red[256];
    float ss = 0.f;
    for (int i = threadIdx.x; i < D_INNER; i += 256) {
        float v = yr[i] * silu_f(zr[i]);
        ss += v * v;
    }
    red[threadIdx.x] = ss; __syncthreads();
    for (int o = 128; o > 0; o >>= 1) {
        if (threadIdx.x < o) red[threadIdx.x] += red[threadIdx.x + o];
        __syncthreads();
    }
    float scale = rsqrtf(red[0] * (1.f / D_INNER) + GNORM_EPS);
    for (int i = threadIdx.x; i < D_INNER; i += 256) {
        float v = yr[i] * silu_f(zr[i]);
        yg[row * D_INNER + i] = (bf16)(v * scale * gw[i]);
    }
}

// out = x + pf*xnorm + pm*(yg @ out_proj_w^T)  (fused epilogue)
__global__ __launch_bounds__(256) void k_gemm_outproj(const bf16* __restrict__ A,
                                                      const bf16* __restrict__ W,
                                                      const float* __restrict__ x,
                                                      const float* __restrict__ xn,
                                                      const float* __restrict__ pfpm,
                                                      float* __restrict__ out) {
    const int wave = threadIdx.x >> 5;
    const int m0 = blockIdx.y * 256 + wave * 32;
    const int n0 = blockIdx.x * 64;
    f32x8 z = {0.f, 0.f, 0.f, 0.f, 0.f, 0.f, 0.f, 0.f};
    f32x8 acc[8] = {z, z, z, z, z, z, z, z};
    mma_tile32x64(A, W, D_INNER, m0, n0, acc);
    const int lane = threadIdx.x & 31;
    const int r = lane & 15, rb = (lane >> 4) * 8;
#pragma unroll
    for (int i = 0; i < 2; i++)
#pragma unroll
        for (int j = 0; j < 4; j++) {
            int col = n0 + j * 16 + r;
            f32x8 v = acc[i * 4 + j];
#pragma unroll
            for (int q = 0; q < 8; q++) {
                int m = m0 + i * 16 + rb + q;
                int b = m >> 11;                       // 2048 rows per batch
                size_t idx = (size_t)m * D_MODEL + col;
                out[idx] = x[idx] + pfpm[b * 2] * xn[idx] + pfpm[b * 2 + 1] * v[q];
            }
        }
}

// ---------------------------------------------------------------- launch
extern "C" void kernel_launch(void* const* d_in, const int* in_sizes, int n_in,
                              void* d_out, int out_size, void* d_ws, size_t ws_size,
                              hipStream_t stream) {
    const float* x        = (const float*)d_in[0];
    const float* norm_w   = (const float*)d_in[1];
    const float* r_w1     = (const float*)d_in[2];
    const float* r_b1     = (const float*)d_in[3];
    const float* r_w2     = (const float*)d_in[4];
    const float* r_b2     = (const float*)d_in[5];
    const float* in_proj  = (const float*)d_in[6];
    const float* conv_w   = (const float*)d_in[7];
    const float* conv_b   = (const float*)d_in[8];
    const float* dt_bias  = (const float*)d_in[9];
    const float* A_log    = (const float*)d_in[10];
    const float* D_param  = (const float*)d_in[11];
    const float* gnorm_w  = (const float*)d_in[12];
    const float* out_proj = (const float*)d_in[13];

    char* ws = (char*)d_ws;
    size_t off = 0;
    float* xnorm  = (float*)(ws + off); off += (size_t)ROWS * D_MODEL * 4;     // 32 MB
    bf16*  xnormb = (bf16*)(ws + off);  off += (size_t)ROWS * D_MODEL * 2;     // 16 MB
    bf16*  w1b    = (bf16*)(ws + off);  off += (size_t)D_IN_PROJ * D_MODEL * 2;// 8.7 MB
    bf16*  w2b    = (bf16*)(ws + off);  off += (size_t)D_MODEL * D_INNER * 2;  // 4 MB
    float* zxbcdt = (float*)(ws + off); off += (size_t)ROWS * D_IN_PROJ * 4;   // 139 MB
    float* xbc    = (float*)(ws + off); off += (size_t)ROWS * CONV_DIM * 4;    // 71 MB
    float* ybuf   = (float*)(ws + off); off += (size_t)ROWS * D_INNER * 4;     // 64 MB
    bf16*  ygb    = (bf16*)(ws + off);  off += (size_t)ROWS * D_INNER * 2;     // 32 MB
    float* mean   = (float*)(ws + off); off += (size_t)BATCH * D_MODEL * 4;
    float* pfpm   = (float*)(ws + off); off += 64;

    const int n_w1 = D_IN_PROJ * D_MODEL;   // 4,358,144
    const int n_w2 = D_MODEL * D_INNER;     // 2,097,152
    k_f32_to_bf16<<<(n_w1 + 255) / 256, 256, 0, stream>>>(in_proj, w1b, n_w1);
    k_f32_to_bf16<<<(n_w2 + 255) / 256, 256, 0, stream>>>(out_proj, w2b, n_w2);

    k_rmsnorm<<<ROWS, 256, 0, stream>>>(x, norm_w, xnorm, xnormb);
    k_meanpool<<<dim3(D_MODEL / 256, BATCH), 256, 0, stream>>>(xnorm, mean);
    k_router<<<1, BATCH * 32, 0, stream>>>(mean, r_w1, r_b1, r_w2, r_b2, pfpm);

    // N blocks: ceil(4256/64) = 67 (last tile clamps/overlaps)
    k_gemm_inproj<<<dim3(67, ROWS / 256), 256, 0, stream>>>(xnormb, w1b, zxbcdt);

    k_conv<<<(unsigned)(((size_t)ROWS * CONV_DIM) / 256), 256, 0, stream>>>(zxbcdt, conv_w, conv_b, xbc);

    k_scan<<<BATCH * NHEADS, 256, 0, stream>>>(zxbcdt, xbc, dt_bias, A_log, D_param, ybuf);

    k_gnorm<<<ROWS, 256, 0, stream>>>(ybuf, zxbcdt, gnorm_w, ygb);

    k_gemm_outproj<<<dim3(D_MODEL / 64, ROWS / 256), 256, 0, stream>>>(
        ygb, w2b, x, xnorm, pfpm, (float*)d_out);
}
